// FWMemory_24807731101746
// MI455X (gfx1250) — matrix operand order repaired and checked by reading, not compiled, not run on hardware
//
#include <hip/hip_runtime.h>
#include <hip/hip_bf16.h>
#include <math.h>

typedef __attribute__((ext_vector_type(16))) __bf16 v16bf;
typedef __attribute__((ext_vector_type(8)))  float  v8f;

#define T_STEPS 1024
#define D_IN    2048
#define S_DIM   1024
#define O_DIM   512
#define M_DIM   48
#define M2      (M_DIM*M_DIM)        // 2304
#define X_DIM   4096                 // D + 2*O + S
#define G_DIM   4096                 // 4*S
#define NTILES  (G_DIM/16)           // 256
#define KTILES  (X_DIM/32)           // 128
#define NWG     128
#define NTHR    256
#define KSLICES 4                    // (NWG*8 waves)/NTILES
#define KT_PER  (KTILES/KSLICES)     // 32
#define XS_U    (KT_PER*16)          // 512 uints staged per block

__device__ __forceinline__ float sigm(float x){ return 1.f/(1.f+expf(-x)); }

// ---------------- grid-wide barrier (persistent kernel) ----------------
__device__ __forceinline__ void grid_sync(unsigned* cnt, unsigned* gen, unsigned nwg){
  __syncthreads();
  if (threadIdx.x == 0){
    __threadfence();
    unsigned g = __hip_atomic_load(gen, __ATOMIC_RELAXED, __HIP_MEMORY_SCOPE_AGENT);
    unsigned prev = __hip_atomic_fetch_add(cnt, 1u, __ATOMIC_ACQ_REL, __HIP_MEMORY_SCOPE_AGENT);
    if (prev == nwg - 1u){
      __hip_atomic_store(cnt, 0u, __ATOMIC_RELAXED, __HIP_MEMORY_SCOPE_AGENT);
      __hip_atomic_fetch_add(gen, 1u, __ATOMIC_RELEASE, __HIP_MEMORY_SCOPE_AGENT);
    } else {
      while (__hip_atomic_load(gen, __ATOMIC_ACQUIRE, __HIP_MEMORY_SCOPE_AGENT) == g){
        __builtin_amdgcn_s_sleep(2);
      }
    }
    __threadfence();
  }
  __syncthreads();
}

// ---------------- weight tiling: fp32 row-major -> bf16 WMMA B-operand tiles ----
// Dest element d = (((nt*KTILES + kt)*32 + lane)*16 + j)
//   lane <16 : N = nt*16+lane,     K = kt*32 + j + 8*(j>=8)
//   lane>=16 : N = nt*16+lane-16,  K = kt*32 + 8 + j + 8*(j>=8)
__global__ void prep_wt(const float* __restrict__ W, __bf16* __restrict__ Wt){
  unsigned d  = blockIdx.x*NTHR + threadIdx.x;      // < 2^24
  unsigned j  = d & 15u;
  unsigned l  = (d>>4) & 31u;
  unsigned kt = (d>>9) & (KTILES-1u);
  unsigned nt = d >> 16;
  unsigned k  = kt*32u + ((j>=8u)?8u:0u) + ((l>=16u)?8u:0u) + j;
  unsigned n  = nt*16u + (l & 15u);
  Wt[d] = (__bf16)W[(size_t)k*G_DIM + n];
}

// ---------------- state init ----------------
__global__ void init_state(const float* __restrict__ inputs, float* __restrict__ memory,
                           float* __restrict__ hbuf, float* __restrict__ cbuf,
                           __bf16* __restrict__ xbf, unsigned* __restrict__ bar){
  unsigned i = blockIdx.x*NTHR + threadIdx.x;
  if (i < (unsigned)(M_DIM*M2)) memory[i] = 0.f;
  if (i < S_DIM){ hbuf[i] = 0.f; cbuf[i] = 0.f; }
  if (i < X_DIM) xbf[i] = (i < D_IN) ? (__bf16)inputs[i] : (__bf16)0.f;
  if (i < 2) bar[i] = 0u;
}

// ---------------- persistent recurrent kernel ----------------
__global__ __launch_bounds__(NTHR) void fw_main(
    const float* __restrict__ inputs, const float* __restrict__ labels,
    const float* __restrict__ b_lstm,
    const float* __restrict__ W_write, const float* __restrict__ b_write,
    const float* __restrict__ W_read,  const float* __restrict__ b_read,
    const float* __restrict__ W_rproj, const float* __restrict__ b_rproj,
    const float* __restrict__ W_out,   const float* __restrict__ b_out,
    const __bf16* __restrict__ Wt,
    float* __restrict__ gatesP, float* __restrict__ hbuf, float* __restrict__ cbuf,
    float* __restrict__ memory, __bf16* __restrict__ xbf,
    unsigned* __restrict__ bar, float* __restrict__ out)
{
  __shared__ __align__(16) unsigned xlds[XS_U];   // this block's x K-slice (2KB)
  __shared__ float hlds[S_DIM];
  __shared__ float keyv[M2];
  __shared__ float uvec[S_DIM];
  __shared__ float errS[O_DIM];
  __shared__ float wr[3*M_DIM+1];
  __shared__ float rd[2*M_DIM];
  __shared__ float k1s[M_DIM],k2s[M_DIM],vs[M_DIM],ns[M_DIM],es[M_DIM];
  __shared__ float voldS[M_DIM], nvecS[M_DIM], nrmS[M_DIM];
  __shared__ float red8[8];
  __shared__ float betaS, invnormS;

  const int tid  = threadIdx.x;
  const int lane = tid & 31;
  const int wv   = tid >> 5;
  const int gw   = blockIdx.x*(NTHR/32) + wv;   // 0..1023
  const int nt   = gw & (NTILES-1);
  const int ks   = gw >> 8;                     // 0..3 (uniform within a block)
  unsigned* cnt = bar; unsigned* gen = bar+1;
  const unsigned* xg = (const unsigned*)xbf + (size_t)ks*XS_U;
  const v16bf* wp = ((const v16bf*)Wt) + ((size_t)nt*KTILES + ks*KT_PER)*32 + lane;
  // All lanes fetch the row-0 pattern: lanes 0-15 the lane-0 halves, 16-31 the
  // lane-16 halves. Rows 1-15 of A become finite duplicates of row 0 -- we only
  // consume D row 0, so no masking/zeroing is needed (branch-free inner loop).
  const int abase = (lane & 16) ? 4 : 0;

  for (int t = 0; t < T_STEPS; ++t){
    // ---- stage this block's x slice (bf16 pairs) into LDS ----
    for (int i = tid; i < XS_U; i += NTHR) xlds[i] = xg[i];
    __syncthreads();

    // ---- phase 1: gates = x @ W_lstm via WMMA (M=1 row in 16x16 tile) ----
    v8f acc = {};
    #pragma unroll 4
    for (int q = 0; q < KT_PER; ++q){
      union { v16bf v; unsigned u[8]; } A;
      #pragma unroll
      for (int i = 0; i < 4; ++i){
        A.u[i]   = xlds[q*16 + abase + i];
        A.u[4+i] = xlds[q*16 + 8 + abase + i];
      }
      v16bf B = wp[(size_t)q*32];             // one contiguous 32B load per lane
      acc = __builtin_amdgcn_wmma_f32_16x16x32_bf16(false, A.v, false, B,
                                                    (short)0, acc, false, false);
    }
    // D row 0 lives in lanes 0..15, VGPR0
    if (lane < 16) gatesP[ks*G_DIM + nt*16 + lane] = acc[0];

    grid_sync(cnt, gen, NWG);

    // ---- phase 2: everything else, one workgroup ----
    if (blockIdx.x == 0){
      // (a) LSTM pointwise
      for (int j = tid; j < S_DIM; j += NTHR){
        float ig = b_lstm[j], gg = b_lstm[S_DIM+j], fg = b_lstm[2*S_DIM+j], og = b_lstm[3*S_DIM+j];
        #pragma unroll
        for (int s = 0; s < KSLICES; ++s){
          const float* gp = gatesP + (size_t)s*G_DIM;
          ig += gp[j]; gg += gp[S_DIM+j]; fg += gp[2*S_DIM+j]; og += gp[3*S_DIM+j];
        }
        float cc = cbuf[j];
        cc = sigm(fg + 1.f)*cc + sigm(ig)*tanhf(gg);
        float hh = sigm(og)*tanhf(cc);
        cbuf[j] = cc; hbuf[j] = hh; hlds[j] = hh;
      }
      __syncthreads();
      // (b) write/read GEMVs (241 outputs, reduce over 1024)
      {
        float a0 = 0.f;
        if (tid < 3*M_DIM+1){
          const float* wq = W_write + tid;
          for (int j = 0; j < S_DIM; ++j) a0 += hlds[j]*wq[(size_t)j*(3*M_DIM+1)];
          wr[tid] = a0 + b_write[tid];
        } else if (tid < (3*M_DIM+1)+2*M_DIM){
          int n = tid - (3*M_DIM+1);
          const float* rp = W_read + n;
          for (int j = 0; j < S_DIM; ++j) a0 += hlds[j]*rp[(size_t)j*(2*M_DIM)];
          rd[n] = a0 + b_read[n];
        }
      }
      __syncthreads();
      // (c) nonlinearities
      if (tid < M_DIM){
        k1s[tid] = tanhf(wr[tid]);       k2s[tid] = tanhf(wr[M_DIM+tid]);
        vs[tid]  = tanhf(wr[2*M_DIM+tid]);
        ns[tid]  = tanhf(rd[tid]);       es[tid]  = tanhf(rd[M_DIM+tid]);
      }
      if (tid == 0) betaS = sigm(wr[3*M_DIM]);
      __syncthreads();
      for (int m = tid; m < M2; m += NTHR) keyv[m] = k1s[m/M_DIM]*k2s[m%M_DIM];
      __syncthreads();
      // (d) v_old = memory @ key_vec
      for (int r = wv; r < M_DIM; r += 8){
        float s = 0.f; const float* mr = memory + (size_t)r*M2;
        for (int m = lane; m < M2; m += 32) s += mr[m]*keyv[m];
        #pragma unroll
        for (int o = 16; o; o >>= 1) s += __shfl_down(s, o, 32);
        if (lane == 0) voldS[r] = s;
      }
      __syncthreads();
      // (e) rank-1 memory update + Frobenius renorm
      {
        float beta = betaS, ssq = 0.f;
        for (int idx = tid; idx < M_DIM*M2; idx += NTHR){
          int r = idx / M2, m = idx - r*M2;
          float nv = memory[idx] + beta*(vs[r]-voldS[r])*keyv[m];
          memory[idx] = nv; ssq += nv*nv;
        }
        #pragma unroll
        for (int o = 16; o; o >>= 1) ssq += __shfl_down(ssq, o, 32);
        if (lane == 0) red8[wv] = ssq;
        __syncthreads();
        if (tid == 0){
          float tt = 0.f;
          for (int i = 0; i < 8; ++i) tt += red8[i];
          invnormS = 1.f/fmaxf(1.f, sqrtf(tt));
        }
        __syncthreads();
        float inv = invnormS;
        for (int idx = tid; idx < M_DIM*M2; idx += NTHR) memory[idx] *= inv;
      }
      __syncthreads();
      // (f) read-key, nvec, LayerNorm(48)
      for (int m = tid; m < M2; m += NTHR) keyv[m] = ns[m/M_DIM]*es[m%M_DIM];
      __syncthreads();
      for (int r = wv; r < M_DIM; r += 8){
        float s = 0.f; const float* mr = memory + (size_t)r*M2;
        for (int m = lane; m < M2; m += 32) s += mr[m]*keyv[m];
        #pragma unroll
        for (int o = 16; o; o >>= 1) s += __shfl_down(s, o, 32);
        if (lane == 0) nvecS[r] = s;
      }
      __syncthreads();
      if (tid == 0){
        float mu = 0.f; for (int i = 0; i < M_DIM; ++i) mu += nvecS[i]; mu *= (1.f/M_DIM);
        float va = 0.f; for (int i = 0; i < M_DIM; ++i){ float dd = nvecS[i]-mu; va += dd*dd; }
        va *= (1.f/M_DIM);
        float isd = rsqrtf(va + 1e-5f);
        for (int i = 0; i < M_DIM; ++i) nrmS[i] = (nvecS[i]-mu)*isd;
      }
      __syncthreads();
      // (g) readout + residual
      for (int j = tid; j < S_DIM; j += NTHR){
        float s = b_rproj[j];
        #pragma unroll 8
        for (int r = 0; r < M_DIM; ++r) s += nrmS[r]*W_rproj[(size_t)r*S_DIM + j];
        uvec[j] = hlds[j] + s;
      }
      __syncthreads();
      // (h) output GEMV + tanh bound + error
      {
        int n0 = tid, n1 = tid + NTHR;
        float o0 = b_out[n0], o1 = b_out[n1];
        #pragma unroll 4
        for (int j = 0; j < S_DIM; ++j){
          float u = uvec[j];
          o0 += u*W_out[(size_t)j*O_DIM + n0];
          o1 += u*W_out[(size_t)j*O_DIM + n1];
        }
        o0 = tanhf(o0*0.1f)*10.f;
        o1 = tanhf(o1*0.1f)*10.f;
        out[(size_t)t*O_DIM + n0] = o0;
        out[(size_t)t*O_DIM + n1] = o1;
        errS[n0] = o0 - labels[(size_t)t*O_DIM + n0];
        errS[n1] = o1 - labels[(size_t)t*O_DIM + n1];
      }
      __syncthreads();
      // (j) build x_{t+1} = [inp_{t+1}, error_t, label_t, h_t] in bf16
      if (t + 1 < T_STEPS){
        const float* inp1 = inputs + (size_t)(t+1)*D_IN;
        for (int k = tid; k < X_DIM; k += NTHR){
          float xv;
          if      (k < D_IN)           xv = inp1[k];
          else if (k < D_IN + O_DIM)   xv = errS[k - D_IN];
          else if (k < D_IN + 2*O_DIM) xv = labels[(size_t)t*O_DIM + (k - (D_IN + O_DIM))];
          else                         xv = hlds[k - (D_IN + 2*O_DIM)];
          xbf[k] = (__bf16)xv;
        }
      }
    }
    grid_sync(cnt, gen, NWG);
  }
}

extern "C" void kernel_launch(void* const* d_in, const int* in_sizes, int n_in,
                              void* d_out, int out_size, void* d_ws, size_t ws_size,
                              hipStream_t stream){
  const float* inputs  = (const float*)d_in[0];
  const float* labels  = (const float*)d_in[1];
  const float* W_lstm  = (const float*)d_in[2];
  const float* b_lstm  = (const float*)d_in[3];
  const float* W_write = (const float*)d_in[4];
  const float* b_write = (const float*)d_in[5];
  const float* W_read  = (const float*)d_in[6];
  const float* b_read  = (const float*)d_in[7];
  const float* W_rproj = (const float*)d_in[8];
  const float* b_rproj = (const float*)d_in[9];
  const float* W_out   = (const float*)d_in[10];
  const float* b_out   = (const float*)d_in[11];

  char* ws = (char*)d_ws;
  size_t off = 0;
  __bf16*   Wt     = (__bf16*)(ws + off);   off += (size_t)G_DIM*X_DIM*2;   // 32 MiB
  float*    memory = (float*)(ws + off);    off += (size_t)M_DIM*M2*4;      // 432 KiB
  float*    gatesP = (float*)(ws + off);    off += (size_t)KSLICES*G_DIM*4; // 64 KiB
  float*    hbuf   = (float*)(ws + off);    off += (size_t)S_DIM*4;
  float*    cbuf   = (float*)(ws + off);    off += (size_t)S_DIM*4;
  __bf16*   xbf    = (__bf16*)(ws + off);   off += (size_t)X_DIM*2;
  unsigned* bar    = (unsigned*)(ws + off); off += 256;

  prep_wt<<<(G_DIM*X_DIM)/NTHR, NTHR, 0, stream>>>(W_lstm, Wt);
  init_state<<<(M_DIM*M2 + NTHR - 1)/NTHR, NTHR, 0, stream>>>(inputs, memory, hbuf, cbuf, xbf, bar);
  fw_main<<<NWG, NTHR, 0, stream>>>(inputs, labels, b_lstm, W_write, b_write,
                                    W_read, b_read, W_rproj, b_rproj, W_out, b_out,
                                    Wt, gatesP, hbuf, cbuf, memory, xbf, bar, (float*)d_out);
}
